// GAT_LSTM_71373766525385
// MI455X (gfx1250) — compile-verified, matrix operand order checked by reading, not run on hardware
//
#include <hip/hip_runtime.h>
#include <hip/hip_bf16.h>

typedef __attribute__((ext_vector_type(16))) _Float16 v16h;
typedef __attribute__((ext_vector_type(8)))  float    v8f;
typedef __attribute__((ext_vector_type(4)))  unsigned int u32x4;
typedef __attribute__((ext_vector_type(8)))  int      i32x8;
typedef __attribute__((ext_vector_type(4)))  int      i32x4;

#define T_DIM 12
#define N_NODES 10000
#define F_IN 64
#define HEADS 4
#define CDIM 32
#define HC 128           // HEADS*CDIM
#define HID 32
#define E_EDGES 1000000
#define TN (T_DIM * N_NODES)          // 120000
#define NE_TOT (E_EDGES + TN)         // 1120000

// ---------------------------------------------------------------- helpers

// native v_exp_f32 + v_rcp_f32, no IEEE div expansion
__device__ __forceinline__ float sigf(float x) {
    return __builtin_amdgcn_rcpf(1.0f + __expf(-x));
}
// branch-free tanh; exact limits at +-inf
__device__ __forceinline__ float tanh_fast(float x) {
    return 1.0f - 2.0f * __builtin_amdgcn_rcpf(__expf(2.0f * x) + 1.0f);
}

__device__ __forceinline__ void atomicMaxF(float* addr, float val) {
    int iv = __float_as_int(val);
    if (iv >= 0) atomicMax((int*)addr, iv);
    else         atomicMin((unsigned int*)addr, (unsigned int)iv);
}

// Issue a TDM 2D tile load (rows x cols of f32, contiguous) global -> LDS, wave-level.
__device__ __forceinline__ void tdm_load_f32(void* lds_dst, const void* gsrc, int cols, int rows) {
    unsigned lds_off = (unsigned)(size_t)lds_dst;           // addr[31:0] = LDS offset
    unsigned long long ga = (unsigned long long)(size_t)gsrc;
    u32x4 g0;
    g0[0] = 1u;                                             // count=1, user mode
    g0[1] = lds_off;                                        // bits 63:32 lds_addr
    g0[2] = (unsigned)(ga & 0xFFFFFFFFu);                   // global_addr[31:0]
    g0[3] = (unsigned)((ga >> 32) & 0x1FFFFFFu) | (2u << 30); // addr[56:32] | type=2
    i32x8 g1;
    g1[0] = (2 << 16);                                      // data_size=2 (4 bytes)
    g1[1] = (cols & 0xFFFF) << 16;                          // tensor_dim0[15:0] @ bits 63:48
    g1[2] = ((cols >> 16) & 0xFFFF) | ((rows & 0xFFFF) << 16); // dim0 hi | tensor_dim1 lo
    g1[3] = ((rows >> 16) & 0xFFFF) | ((cols & 0xFFFF) << 16); // dim1 hi | tile_dim0
    g1[4] = rows & 0xFFFF;                                  // tile_dim1 (tile_dim2=0)
    g1[5] = cols;                                           // tensor_dim0_stride[31:0]
    g1[6] = 0;
    g1[7] = 0;
    i32x4 z4 = {0, 0, 0, 0};
    i32x8 z8 = {0, 0, 0, 0, 0, 0, 0, 0};
#if __has_builtin(__builtin_amdgcn_tensor_load_to_lds)
    __builtin_amdgcn_tensor_load_to_lds(g0, g1, z4, z4, z8, 0);
    __builtin_amdgcn_s_wait_tensorcnt(0);
#endif
}

// ---------------------------------------------------------------- init

__global__ __launch_bounds__(256) void init_kernel(float* agg, float* mbuf, float* dbuf, float* scal) {
    long long idx = (long long)blockIdx.x * 256 + threadIdx.x;
    if (idx < (long long)TN * HC) agg[idx] = 0.0f;
    if (idx < (long long)TN * HEADS) { mbuf[idx] = -__builtin_inff(); dbuf[idx] = 0.0f; }
    if (idx < 8) scal[idx] = 0.0f;
}

// ---------------------------------------------------------------- WMMA GEMM  C[M x 128] = A[M x K] * B + bias
// TRANSB==0: W stored [K][128] row-major (B = W), staged to LDS via TDM
// TRANSB==1: W stored [128][K] row-major (B = W^T), staged transposed
// PERMC==1: output row r=(t*N+n) is stored at [n][t][128] for the scan's streaming reads.

template<int TRANSB, int K, int PERMC>
__global__ __launch_bounds__(256) void gemm_wmma_kernel(
        const float* __restrict__ A, const float* __restrict__ W, float* __restrict__ C,
        const float* __restrict__ bias0, const float* __restrict__ bias1) {
    __shared__ __attribute__((aligned(16))) float wlds[K * HC];   // B tile [k][n] f32 (<= 64 KB)

    const int wave  = threadIdx.x >> 5;
    const int lane  = threadIdx.x & 31;
    const int row   = blockIdx.x * 16 + (lane & 15);
    const int ncol  = wave * 16 + (lane & 15);
    const int khalf = (lane < 16) ? 0 : 8;

    // ---- stage B into LDS, layout [k][n] ----
    if (TRANSB) {
#pragma unroll
        for (int i = threadIdx.x * 4; i < HC * K; i += 256 * 4) {
            int n = i / K, k = i % K;           // 4 consecutive k, constant divisor
            float4 v = *(const float4*)(W + i);
            wlds[(k + 0) * HC + n] = v.x;
            wlds[(k + 1) * HC + n] = v.y;
            wlds[(k + 2) * HC + n] = v.z;
            wlds[(k + 3) * HC + n] = v.w;
        }
    } else {
        if (threadIdx.x < 32) tdm_load_f32(wlds, W, HC, K);   // async tensor DMA, wave 0
    }
    __syncthreads();

    v8f acc = {};
    const float* arow = A + (size_t)row * K;
#pragma unroll
    for (int kk = 0; kk < K; kk += 32) {
        // A fragment: two contiguous 8-float runs -> 4x global_load_b128
        float4 a0 = *(const float4*)(arow + kk + khalf);
        float4 a1 = *(const float4*)(arow + kk + khalf + 4);
        float4 a2 = *(const float4*)(arow + kk + khalf + 16);
        float4 a3 = *(const float4*)(arow + kk + khalf + 20);
        v16h a;
        a[0]  = (_Float16)a0.x; a[1]  = (_Float16)a0.y; a[2]  = (_Float16)a0.z; a[3]  = (_Float16)a0.w;
        a[4]  = (_Float16)a1.x; a[5]  = (_Float16)a1.y; a[6]  = (_Float16)a1.z; a[7]  = (_Float16)a1.w;
        a[8]  = (_Float16)a2.x; a[9]  = (_Float16)a2.y; a[10] = (_Float16)a2.z; a[11] = (_Float16)a2.w;
        a[12] = (_Float16)a3.x; a[13] = (_Float16)a3.y; a[14] = (_Float16)a3.z; a[15] = (_Float16)a3.w;
        v16h b;
#pragma unroll
        for (int e = 0; e < 16; ++e) {
            int k = kk + khalf + ((e < 8) ? e : (e + 8));
            b[e] = (_Float16)wlds[k * HC + ncol];
        }
        acc = __builtin_amdgcn_wmma_f32_16x16x32_f16(false, a, false, b, (short)0, acc, false, false);
    }
    float bias = 0.0f;
    if (bias0) bias += bias0[ncol];
    if (bias1) bias += bias1[ncol];
#pragma unroll
    for (int r = 0; r < 8; ++r) {
        int orow = blockIdx.x * 16 + r + ((lane < 16) ? 0 : 8);
        if (PERMC) {
            int t  = orow / N_NODES;            // each 16-row block lies in one t (10000 % 16 == 0)
            int nn = orow - t * N_NODES;
            C[(size_t)nn * (T_DIM * HC) + t * HC + ncol] = acc[r] + bias;
        } else {
            C[(size_t)orow * HC + ncol] = acc[r] + bias;
        }
    }
}

// ---------------------------------------------------------------- per-node attention logits

__global__ __launch_bounds__(256) void att_kernel(const float* __restrict__ hfeat,
        const float* __restrict__ att_src, const float* __restrict__ att_dst,
        float* __restrict__ asrc, float* __restrict__ adst) {
    int tid = blockIdx.x * 256 + threadIdx.x;
    if (tid >= TN * HEADS) return;
    int n = tid >> 2, hh = tid & 3;
    const float* hp = hfeat + (size_t)n * HC + hh * CDIM;
    float s = 0.0f, d = 0.0f;
#pragma unroll
    for (int c = 0; c < CDIM; c += 4) {
        float4 v = *(const float4*)(hp + c);
        s += v.x * att_src[hh * CDIM + c]     + v.y * att_src[hh * CDIM + c + 1]
           + v.z * att_src[hh * CDIM + c + 2] + v.w * att_src[hh * CDIM + c + 3];
        d += v.x * att_dst[hh * CDIM + c]     + v.y * att_dst[hh * CDIM + c + 1]
           + v.z * att_dst[hh * CDIM + c + 2] + v.w * att_dst[hh * CDIM + c + 3];
    }
    asrc[tid] = s;
    adst[tid] = d;
}

// ---------------------------------------------------------------- edge_attr mean reduction + ke[h]

__global__ __launch_bounds__(256) void easum_kernel(const float* __restrict__ ea, float* scal) {
    __shared__ float red[256];
    float s = 0.0f;
    for (long long i = (long long)blockIdx.x * 256 + threadIdx.x; i < E_EDGES; i += (long long)gridDim.x * 256)
        s += ea[i];
    red[threadIdx.x] = s;
    __syncthreads();
    for (int st = 128; st > 0; st >>= 1) {
        if ((int)threadIdx.x < st) red[threadIdx.x] += red[threadIdx.x + st];
        __syncthreads();
    }
    if (threadIdx.x == 0) atomicAdd(&scal[0], red[0]);
}

__global__ void prep_kernel(const float* __restrict__ W_edge, const float* __restrict__ att_edge,
                            float* scal) {
    int t = threadIdx.x;
    if (t == 4) scal[1] = scal[0] / (float)E_EDGES;           // ea_mean
    if (t < 4) {                                              // ke[h]
        float s = 0.0f;
        for (int c = 0; c < CDIM; ++c) s += W_edge[t * CDIM + c] * att_edge[t * CDIM + c];
        scal[2 + t] = s;
    }
}

// ---------------------------------------------------------------- edge passes

__device__ __forceinline__ void edge_info(int e, const int* __restrict__ ei,
        const float* __restrict__ ea, const float* __restrict__ scal,
        int& src, int& dst, float& av) {
    if (e < E_EDGES) { src = ei[e]; dst = ei[E_EDGES + e]; av = ea[e]; }
    else             { src = dst = e - E_EDGES; av = scal[1]; }
}

__global__ __launch_bounds__(256) void edge_max_kernel(const int* __restrict__ ei,
        const float* __restrict__ ea, const float* __restrict__ asrc, const float* __restrict__ adst,
        const float* __restrict__ scal, float* __restrict__ mbuf) {
    int e = blockIdx.x * 256 + threadIdx.x;
    if (e >= NE_TOT) return;
    int src, dst; float av;
    edge_info(e, ei, ea, scal, src, dst, av);
#pragma unroll
    for (int hh = 0; hh < HEADS; ++hh) {
        float al = asrc[src * HEADS + hh] + adst[dst * HEADS + hh] + scal[2 + hh] * av;
        al = (al > 0.0f) ? al : 0.2f * al;
        atomicMaxF(&mbuf[dst * HEADS + hh], al);
    }
}

__global__ __launch_bounds__(256) void edge_sum_kernel(const int* __restrict__ ei,
        const float* __restrict__ ea, const float* __restrict__ asrc, const float* __restrict__ adst,
        const float* __restrict__ scal, const float* __restrict__ mbuf, float* __restrict__ dbuf) {
    int e = blockIdx.x * 256 + threadIdx.x;
    if (e >= NE_TOT) return;
    int src, dst; float av;
    edge_info(e, ei, ea, scal, src, dst, av);
#pragma unroll
    for (int hh = 0; hh < HEADS; ++hh) {
        float al = asrc[src * HEADS + hh] + adst[dst * HEADS + hh] + scal[2 + hh] * av;
        al = (al > 0.0f) ? al : 0.2f * al;
        atomicAdd(&dbuf[dst * HEADS + hh], __expf(al - mbuf[dst * HEADS + hh]));
    }
}

// 32 lanes per edge: lane l owns head h = l>>3 and 4 contiguous channels
__global__ __launch_bounds__(256) void edge_agg_kernel(const int* __restrict__ ei,
        const float* __restrict__ ea, const float* __restrict__ asrc, const float* __restrict__ adst,
        const float* __restrict__ scal, const float* __restrict__ mbuf, const float* __restrict__ dbuf,
        const float* __restrict__ hfeat, float* __restrict__ agg) {
    long long tid = (long long)blockIdx.x * 256 + threadIdx.x;
    int e = (int)(tid >> 5);
    int l = (int)(tid & 31);
    if (e >= NE_TOT) return;
    int src, dst; float av;
    edge_info(e, ei, ea, scal, src, dst, av);
    int hh = l >> 3;
    int ch = hh * CDIM + (l & 7) * 4;
    float al = asrc[src * HEADS + hh] + adst[dst * HEADS + hh] + scal[2 + hh] * av;
    al = (al > 0.0f) ? al : 0.2f * al;
    float w = __expf(al - mbuf[dst * HEADS + hh]) *
              __builtin_amdgcn_rcpf(dbuf[dst * HEADS + hh] + 1e-16f);
    float4 hv = *(const float4*)(hfeat + (size_t)src * HC + ch);
    float* dp = agg + (size_t)dst * HC + ch;
    atomicAdd(dp + 0, hv.x * w);
    atomicAdd(dp + 1, hv.y * w);
    atomicAdd(dp + 2, hv.z * w);
    atomicAdd(dp + 3, hv.w * w);
}

__global__ __launch_bounds__(256) void bias_relu_kernel(float* __restrict__ agg,
                                                        const float* __restrict__ gat_bias) {
    long long idx = (long long)blockIdx.x * 256 + threadIdx.x;
    if (idx >= (long long)TN * HC) return;
    float v = agg[idx] + gat_bias[idx & (HC - 1)];
    agg[idx] = v > 0.0f ? v : 0.0f;
}

// ---------------------------------------------------------------- sequential LSTM scan over 10000 nodes
// 384 threads = 12 waves. Waves 0..7: one WMMA N-tile each (hh @ W_hh^T, K=32, one instruction).
// Gate phase: exactly one (t,j) item per thread (ti = wave, j = lane), fully coalesced G reads.
// G is pre-permuted to [n][t][128] so each step streams one contiguous 6 KB row.

__global__ __launch_bounds__(384) void lstm_scan_kernel(const float* __restrict__ G,
        const float* __restrict__ W_hh, const float* __restrict__ W_fc,
        const float* __restrict__ b_fc, float* __restrict__ out, int N) {
    __shared__ __attribute__((aligned(16))) float hh_lds[16 * HID];   // rows 12..15 stay zero
    __shared__ float cc_lds[T_DIM * HID];
    __shared__ float g_lds[16 * HC];                                  // rows 12..15 scratch
    __shared__ float wfc[HID];

    const int tid   = threadIdx.x;
    const int wave  = tid >> 5;
    const int lane  = tid & 31;
    const int ncol  = lane & 15;
    const int khalf = (lane < 16) ? 0 : 8;
    const int col   = (wave & 7) * 16 + ncol;
    const int rbase = (lane < 16) ? 0 : 8;

    for (int i = tid; i < 16 * HID; i += 384) hh_lds[i] = 0.0f;
    for (int i = tid; i < T_DIM * HID; i += 384) cc_lds[i] = 0.0f;
    if (tid < HID) wfc[tid] = W_fc[tid];

    // loop-invariant B fragment: B[k][g] = W_hh[g][k], tile columns g = col (waves 0..7 only use it)
    v16h bfrag;
#pragma unroll
    for (int e = 0; e < 16; ++e) {
        int k = khalf + ((e < 8) ? e : (e + 8));
        bfrag[e] = (_Float16)W_hh[col * HID + k];
    }
    __syncthreads();

    const float bfc = b_fc[0];
    for (int n = 0; n < N; ++n) {
        if (wave < 8) {
            // A fragment: 4x ds_load_b128 from padded hidden state (no guards)
            const float* hp = hh_lds + (lane & 15) * HID + khalf;
            float4 a0 = *(const float4*)(hp);
            float4 a1 = *(const float4*)(hp + 4);
            float4 a2 = *(const float4*)(hp + 16);
            float4 a3 = *(const float4*)(hp + 20);
            v16h a;
            a[0]  = (_Float16)a0.x; a[1]  = (_Float16)a0.y; a[2]  = (_Float16)a0.z; a[3]  = (_Float16)a0.w;
            a[4]  = (_Float16)a1.x; a[5]  = (_Float16)a1.y; a[6]  = (_Float16)a1.z; a[7]  = (_Float16)a1.w;
            a[8]  = (_Float16)a2.x; a[9]  = (_Float16)a2.y; a[10] = (_Float16)a2.z; a[11] = (_Float16)a2.w;
            a[12] = (_Float16)a3.x; a[13] = (_Float16)a3.y; a[14] = (_Float16)a3.z; a[15] = (_Float16)a3.w;
            v8f acc = {};
            acc = __builtin_amdgcn_wmma_f32_16x16x32_f16(false, a, false, bfrag, (short)0, acc, false, false);
            // store recurrent contribution (rows 12..15 are dead scratch)
#pragma unroll
            for (int r = 0; r < 8; ++r) g_lds[(r + rbase) * HC + col] = acc[r];
        }
        __syncthreads();

        // gates: recurrent (LDS) + streamed input part (contiguous global row, b_ih+b_hh folded in)
        const float* gn = G + (size_t)n * (T_DIM * HC);
        {
            const float* gr = gn + wave * HC + lane;
            const float* lr = g_lds + wave * HC + lane;
            float gi = lr[0]        + gr[0];
            float gf = lr[CDIM]     + gr[CDIM];
            float gg = lr[2 * CDIM] + gr[2 * CDIM];
            float go = lr[3 * CDIM] + gr[3 * CDIM];
            float c = sigf(gf) * cc_lds[tid] + sigf(gi) * tanh_fast(gg);
            cc_lds[tid] = c;
            hh_lds[tid] = sigf(go) * tanh_fast(c);
        }
        if (tid < 96) __builtin_prefetch(gn + 2 * T_DIM * HC + tid * 16, 0, 3);  // 2 nodes ahead, 96x64B = 6 KB
        __syncthreads();

        // output projection: wave-0 shuffle reduction over hh[11][:]
        if (wave == 0) {
            float s = hh_lds[(T_DIM - 1) * HID + lane] * wfc[lane];
#pragma unroll
            for (int off = 16; off > 0; off >>= 1) s += __shfl_down(s, off, 32);
            if (lane == 0) out[n] = s + bfc;
        }
    }
}

// ---------------------------------------------------------------- launch

extern "C" void kernel_launch(void* const* d_in, const int* in_sizes, int n_in,
                              void* d_out, int out_size, void* d_ws, size_t ws_size,
                              hipStream_t stream) {
    const float* x        = (const float*)d_in[0];
    const int*   ei       = (const int*)  d_in[1];
    const float* ea       = (const float*)d_in[2];
    const float* W_gat    = (const float*)d_in[3];
    const float* att_src  = (const float*)d_in[4];
    const float* att_dst  = (const float*)d_in[5];
    const float* att_edge = (const float*)d_in[6];
    const float* W_edge   = (const float*)d_in[7];
    const float* gat_bias = (const float*)d_in[8];
    const float* W_ih     = (const float*)d_in[9];
    const float* W_hh     = (const float*)d_in[10];
    const float* b_ih     = (const float*)d_in[11];
    const float* b_hh     = (const float*)d_in[12];
    const float* W_fc     = (const float*)d_in[13];
    const float* b_fc     = (const float*)d_in[14];
    float* out = (float*)d_out;

    float* ws = (float*)d_ws;
    size_t off = 0;
    float* h_feat = ws + off; off += (size_t)TN * HC;      // GAT features
    float* agg    = ws + off; off += (size_t)TN * HC;      // aggregated / relu output
    float* G      = ws + off; off += (size_t)TN * HC;      // input gates, permuted [n][t][128]
    float* asrc   = ws + off; off += (size_t)TN * HEADS;
    float* adst   = ws + off; off += (size_t)TN * HEADS;
    float* mbuf   = ws + off; off += (size_t)TN * HEADS;
    float* dbuf   = ws + off; off += (size_t)TN * HEADS;
    float* scal   = ws + off; off += 8;                    // [0]=sum, [1]=mean, [2..5]=ke

    init_kernel<<<(TN * HC + 255) / 256, 256, 0, stream>>>(agg, mbuf, dbuf, scal);

    // h = x @ W_gat        (120000 x 64 @ 64 x 128), B staged by TDM
    gemm_wmma_kernel<0, F_IN, 0><<<TN / 16, 256, 0, stream>>>(x, W_gat, h_feat, nullptr, nullptr);

    att_kernel<<<(TN * HEADS + 255) / 256, 256, 0, stream>>>(h_feat, att_src, att_dst, asrc, adst);
    easum_kernel<<<256, 256, 0, stream>>>(ea, scal);
    prep_kernel<<<1, 64, 0, stream>>>(W_edge, att_edge, scal);

    int eb = (NE_TOT + 255) / 256;
    edge_max_kernel<<<eb, 256, 0, stream>>>(ei, ea, asrc, adst, scal, mbuf);
    edge_sum_kernel<<<eb, 256, 0, stream>>>(ei, ea, asrc, adst, scal, mbuf, dbuf);
    long long agg_threads = (long long)NE_TOT * 32;
    edge_agg_kernel<<<(unsigned)((agg_threads + 255) / 256), 256, 0, stream>>>(
        ei, ea, asrc, adst, scal, mbuf, dbuf, h_feat, agg);

    bias_relu_kernel<<<(TN * HC + 255) / 256, 256, 0, stream>>>(agg, gat_bias);

    // G = relu_out @ W_ih^T + b_ih + b_hh, output permuted to [n][t][128]
    gemm_wmma_kernel<1, HC, 1><<<TN / 16, 256, 0, stream>>>(agg, W_ih, G, b_ih, b_hh);

    // sequential scan over nodes (single workgroup of 12 waves, 8 WMMA tiles per step)
    lstm_scan_kernel<<<1, 384, 0, stream>>>(G, W_hh, W_fc, b_fc, out, N_NODES);
}
